// SimplifiedLinearAttention_35218731827328
// MI455X (gfx1250) — compile-verified, matrix operand order checked by reading, not run on hardware
//
#include <hip/hip_runtime.h>

typedef __bf16 bf16;
typedef __attribute__((ext_vector_type(4)))  bf16  v4bf;
typedef __attribute__((ext_vector_type(8)))  bf16  v8bf;
typedef __attribute__((ext_vector_type(16))) bf16  v16bf;
typedef __attribute__((ext_vector_type(8)))  float v8f;

#define NB   1024
#define NN   196
#define NC   384
#define NH   12
#define HD   32
#define NBH  (NB*NH)              /* 12288 */
#define NM   (NB*NN)              /* 200704 */
#define QE   ((size_t)NBH*NN*HD)  /* 77,070,336 bf16 elems per q/k/v buffer */

__device__ __forceinline__ v8f wmma_bf16(v16bf a, v16bf b, v8f c) {
  // emits v_wmma_f32_16x16x32_bf16
  return __builtin_amdgcn_wmma_f32_16x16x32_bf16(false, a, false, b, (short)0, c,
                                                 false, false);
}

// Load a 16xK=32 A/B fragment from LDS (row-major, K contiguous).
// Per CDNA5 ISA layout: lanes 0-15 hold K=0..7 and 16..23, lanes 16-31 hold
// K=8..15 and 24..31 -> two 16B ds_load_b128 per lane.
__device__ __forceinline__ v16bf frag_ld(const bf16* base, int stride, int row, int k0) {
  const int lane = (int)(threadIdx.x & 31u);
  const int lh   = lane >> 4;
  const bf16* p  = base + row * stride + k0 + lh * 8;
  v8bf lo = *(const v8bf*)p;
  v8bf hi = *(const v8bf*)(p + 16);
  v16bf f;
#pragma unroll
  for (int i = 0; i < 8; ++i) { f[i] = lo[i]; f[i + 8] = hi[i]; }
  return f;
}

// ---------------------------------------------------------------------------
// Kernel 1: qkv = x @ qkv_w^T + b, fused relu / (+pos_enc,relu) / head split.
// Grid (1568, 9), 256 threads. BM=BN=128, BK=32. Output: bf16 q/k/v in
// (b*12+head, i, c) layout.
// ---------------------------------------------------------------------------
__global__ __launch_bounds__(256)
void qkv_gemm_kernel(const float* __restrict__ x, const float* __restrict__ w,
                     const float* __restrict__ bias, const float* __restrict__ pos_enc,
                     bf16* __restrict__ qs, bf16* __restrict__ ks, bf16* __restrict__ vs)
{
  __shared__ __align__(16) bf16 As[128 * 40];
  __shared__ __align__(16) bf16 Bs[128 * 40];

  const int bm   = blockIdx.x;          // 0..1567
  const int bn   = blockIdx.y;          // 0..8
  const int tid  = (int)threadIdx.x;
  const int wave = tid >> 5;
  const int lane = tid & 31;
  const int lh   = lane >> 4;
  const int r    = lane & 15;
  const int wm   = wave >> 1;           // 0..3  (32-row strip)
  const int wn   = wave & 1;            // 0..1  (64-col strip)

  v8f zero = {};
  v8f acc[2][4];
#pragma unroll
  for (int i = 0; i < 2; ++i)
#pragma unroll
    for (int j = 0; j < 4; ++j) acc[i][j] = zero;

  for (int kt = 0; kt < 12; ++kt) {
    const int k0 = kt * 32;
#pragma unroll
    for (int it = 0; it < 4; ++it) {
      const int idx = tid + it * 256;        // 1024 float4-chunks
      const int rr  = idx >> 3;
      const int c4  = (idx & 7) << 2;
      const float4 fa = *(const float4*)(x + (size_t)(bm * 128 + rr) * NC + k0 + c4);
      bf16* da = As + rr * 40 + c4;
      da[0] = (bf16)fa.x; da[1] = (bf16)fa.y; da[2] = (bf16)fa.z; da[3] = (bf16)fa.w;
      const float4 fb = *(const float4*)(w + (size_t)(bn * 128 + rr) * NC + k0 + c4);
      bf16* db = Bs + rr * 40 + c4;
      db[0] = (bf16)fb.x; db[1] = (bf16)fb.y; db[2] = (bf16)fb.z; db[3] = (bf16)fb.w;
    }
    if (kt + 1 < 12) {  // emits global_prefetch_b8 for the next K tile
      __builtin_prefetch(x + (size_t)(bm * 128 + (tid >> 1)) * NC + k0 + 32, 0, 0);
      __builtin_prefetch(w + (size_t)(bn * 128 + (tid >> 1)) * NC + k0 + 32, 0, 0);
    }
    __syncthreads();

    v16bf af[2], bfr[4];
#pragma unroll
    for (int i = 0; i < 2; ++i) af[i]  = frag_ld(As, 40, wm * 32 + i * 16 + r, 0);
#pragma unroll
    for (int j = 0; j < 4; ++j) bfr[j] = frag_ld(Bs, 40, wn * 64 + j * 16 + r, 0);
#pragma unroll
    for (int i = 0; i < 2; ++i)
#pragma unroll
      for (int j = 0; j < 4; ++j)
        acc[i][j] = wmma_bf16(af[i], bfr[j], acc[i][j]);
    __syncthreads();
  }

  // Epilogue: bias, relu / pos_enc+relu, head-split scatter as bf16.
#pragma unroll
  for (int i = 0; i < 2; ++i) {
#pragma unroll
    for (int j = 0; j < 4; ++j) {
      const int n    = bn * 128 + wn * 64 + j * 16 + r;   // 0..1151
      const float bb = bias[n];
      const int sect = n / NC;                            // 0=q 1=k 2=v
      const int cc   = n - sect * NC;                     // 0..383
      const int head = cc >> 5;
      const int c    = cc & 31;
#pragma unroll
      for (int e = 0; e < 8; ++e) {
        const int m  = bm * 128 + wm * 32 + i * 16 + lh * 8 + e;
        const int b_ = m / NN;
        const int i_ = m - b_ * NN;
        float val = acc[i][j][e] + bb;
        const size_t o = ((size_t)(b_ * NH + head) * NN + i_) * HD + c;
        if (sect == 0) {
          qs[o] = (bf16)fmaxf(val, 0.0f);
        } else if (sect == 1) {
          val += pos_enc[i_ * NC + cc];
          ks[o] = (bf16)fmaxf(val, 0.0f);
        } else {
          vs[o] = (bf16)val;
        }
      }
    }
  }
}

// ---------------------------------------------------------------------------
// Kernel 2: per (b,head): kv = k^T v (WMMA), z = 1/(q . ksum), out = q kv * z
// (WMMA), plus fused 5x5 depthwise conv of v, all from LDS. 12288 blocks,
// 128 threads (4 waves).
// ---------------------------------------------------------------------------
__global__ __launch_bounds__(128)
void attn_kernel(const bf16* __restrict__ qs, const bf16* __restrict__ ks,
                 const bf16* __restrict__ vs, const float* __restrict__ dwc_w,
                 const float* __restrict__ dwc_b, bf16* __restrict__ os)
{
  __shared__ __align__(16) bf16 qsm[208 * 32];   // q, rows padded to 208
  __shared__ __align__(16) bf16 kT[32 * 224];    // k^T, j padded to 224
  __shared__ __align__(16) bf16 vT[32 * 224];    // v^T
  __shared__ __align__(16) bf16 kvT[32 * 32];    // kv transposed (d-major)
  __shared__ __align__(16) bf16 outb[196 * 32];  // attention output tile
  __shared__ float ksum[32];
  __shared__ float zsh[208];
  __shared__ float wsm[32 * 25];
  __shared__ float bsm[32];

  const int bh  = (int)blockIdx.x;
  const int tid = (int)threadIdx.x;
  const size_t base = (size_t)bh * (NN * HD);

  // Stage depthwise weights/bias.
  for (int i = tid; i < 800; i += 128) wsm[i] = dwc_w[i];
  if (tid < 32) bsm[tid] = dwc_b[tid];

  // Stage q row-major, k/v transposed (coalesced global reads).
  for (int idx = tid; idx < NN * HD; idx += 128) {
    const int j = idx >> 5, c = idx & 31;
    qsm[j * 32 + c] = qs[base + idx];
    kT[c * 224 + j] = ks[base + idx];
    vT[c * 224 + j] = vs[base + idx];
  }
  for (int idx = tid; idx < 12 * 32; idx += 128) qsm[196 * 32 + idx] = (bf16)0.0f;
  for (int idx = tid; idx < 32 * 28; idx += 128) {
    const int c = idx / 28, j = 196 + (idx - c * 28);
    kT[c * 224 + j] = (bf16)0.0f;
    vT[c * 224 + j] = (bf16)0.0f;
  }
  __syncthreads();

  // ksum[c] = sum_j k[j,c]
  if (tid < 32) {
    float s = 0.0f;
    for (int j = 0; j < NN; ++j) s += (float)kT[tid * 224 + j];
    ksum[tid] = s;
  }

  const int wave = tid >> 5, lane = tid & 31, lh = lane >> 4, r = lane & 15;

  // kv[c,d] = sum_j k[j,c]*v[j,d]: one 16x16 tile per wave, K=224 (7 steps).
  {
    const int cT = wave >> 1, dT = wave & 1;
    v8f acc = {};
    for (int kk = 0; kk < 7; ++kk) {
      v16bf a = frag_ld(kT, 224, cT * 16 + r, kk * 32);
      v16bf b = frag_ld(vT, 224, dT * 16 + r, kk * 32);
      acc = wmma_bf16(a, b, acc);
    }
#pragma unroll
    for (int i = 0; i < 8; ++i) {
      const int c = cT * 16 + lh * 8 + i;
      const int d = dT * 16 + r;
      kvT[d * 32 + c] = (bf16)acc[i];
    }
  }
  __syncthreads();

  // z[i] = 1/(q[i,:].ksum + 1e-6)
  for (int i = tid; i < NN; i += 128) {
    float s = 0.0f;
#pragma unroll
    for (int c = 0; c < 32; ++c) s += (float)qsm[i * 32 + c] * ksum[c];
    zsh[i] = 1.0f / (s + 1e-6f);
  }
  __syncthreads();

  // out = (q @ kv) * z : 13x2 = 26 16x16 tiles over 4 waves, K=32 each.
  for (int t = wave; t < 26; t += 4) {
    const int mt = t >> 1, nt = t & 1;
    v16bf a = frag_ld(qsm, 32, mt * 16 + r, 0);
    v16bf b = frag_ld(kvT, 32, nt * 16 + r, 0);
    v8f o = {};
    o = wmma_bf16(a, b, o);
#pragma unroll
    for (int i = 0; i < 8; ++i) {
      const int gi = mt * 16 + lh * 8 + i;
      if (gi < NN) outb[gi * 32 + nt * 16 + r] = (bf16)(o[i] * zsh[gi]);
    }
  }
  __syncthreads();

  // Fused 5x5 depthwise conv on v (14x14 spatial), add, store bf16.
  for (int idx = tid; idx < NN * HD; idx += 128) {
    const int c = idx & 31, i = idx >> 5;
    const int py = i / 14, px = i - py * 14;
    float a = (float)outb[i * 32 + c] + bsm[c];
#pragma unroll
    for (int dy = 0; dy < 5; ++dy) {
      const int yy = py + dy - 2;
      if (yy < 0 || yy >= 14) continue;
#pragma unroll
      for (int dx = 0; dx < 5; ++dx) {
        const int xx = px + dx - 2;
        if (xx < 0 || xx >= 14) continue;
        a += wsm[c * 25 + dy * 5 + dx] * (float)vT[c * 224 + yy * 14 + xx];
      }
    }
    os[base + idx] = (bf16)a;
  }
}

// ---------------------------------------------------------------------------
// Kernel 3: out = merge_heads(attn_out) @ proj_w^T + proj_b (fp32 output).
// Grid (1568, 3), 256 threads. BK=32 == head dim, so the head-merge gather is
// a contiguous 64B row read per K tile.
// ---------------------------------------------------------------------------
__global__ __launch_bounds__(256)
void proj_gemm_kernel(const bf16* __restrict__ os, const float* __restrict__ w,
                      const float* __restrict__ bias, float* __restrict__ out)
{
  __shared__ __align__(16) bf16 As[128 * 40];
  __shared__ __align__(16) bf16 Bs[128 * 40];

  const int bm   = blockIdx.x;
  const int bn   = blockIdx.y;
  const int tid  = (int)threadIdx.x;
  const int wave = tid >> 5;
  const int lane = tid & 31;
  const int lh   = lane >> 4;
  const int r    = lane & 15;
  const int wm   = wave >> 1;
  const int wn   = wave & 1;

  v8f zero = {};
  v8f acc[2][4];
#pragma unroll
  for (int i = 0; i < 2; ++i)
#pragma unroll
    for (int j = 0; j < 4; ++j) acc[i][j] = zero;

  for (int kt = 0; kt < 12; ++kt) {      // kt == head index
    const int k0 = kt * 32;
#pragma unroll
    for (int it = 0; it < 4; ++it) {
      const int idx = tid + it * 256;
      const int rr  = idx >> 3;
      const int c4  = (idx & 7) << 2;
      const int m   = bm * 128 + rr;
      const int b_  = m / NN;
      const int i_  = m - b_ * NN;
      const size_t abase = ((size_t)(b_ * NH + kt) * NN + i_) * HD + c4;
      const v4bf av = *(const v4bf*)(os + abase);
      bf16* da = As + rr * 40 + c4;
      da[0] = av[0]; da[1] = av[1]; da[2] = av[2]; da[3] = av[3];
      const float4 fb = *(const float4*)(w + (size_t)(bn * 128 + rr) * NC + k0 + c4);
      bf16* db = Bs + rr * 40 + c4;
      db[0] = (bf16)fb.x; db[1] = (bf16)fb.y; db[2] = (bf16)fb.z; db[3] = (bf16)fb.w;
    }
    if (kt + 1 < 12)
      __builtin_prefetch(w + (size_t)(bn * 128 + (tid >> 1)) * NC + k0 + 32, 0, 0);
    __syncthreads();

    v16bf af[2], bfr[4];
#pragma unroll
    for (int i = 0; i < 2; ++i) af[i]  = frag_ld(As, 40, wm * 32 + i * 16 + r, 0);
#pragma unroll
    for (int j = 0; j < 4; ++j) bfr[j] = frag_ld(Bs, 40, wn * 64 + j * 16 + r, 0);
#pragma unroll
    for (int i = 0; i < 2; ++i)
#pragma unroll
      for (int j = 0; j < 4; ++j)
        acc[i][j] = wmma_bf16(af[i], bfr[j], acc[i][j]);
    __syncthreads();
  }

#pragma unroll
  for (int i = 0; i < 2; ++i) {
#pragma unroll
    for (int j = 0; j < 4; ++j) {
      const int n    = bn * 128 + wn * 64 + j * 16 + r;  // 0..383
      const float bb = bias[n];
#pragma unroll
      for (int e = 0; e < 8; ++e) {
        const int m = bm * 128 + wm * 32 + i * 16 + lh * 8 + e;
        out[(size_t)m * NC + n] = acc[i][j][e] + bb;
      }
    }
  }
}

// ---------------------------------------------------------------------------
extern "C" void kernel_launch(void* const* d_in, const int* in_sizes, int n_in,
                              void* d_out, int out_size, void* d_ws, size_t ws_size,
                              hipStream_t stream) {
  (void)in_sizes; (void)n_in; (void)out_size; (void)ws_size;
  const float* x       = (const float*)d_in[0];
  const float* qkv_w   = (const float*)d_in[1];
  const float* qkv_b   = (const float*)d_in[2];
  const float* proj_w  = (const float*)d_in[3];
  const float* proj_b  = (const float*)d_in[4];
  const float* dwc_w   = (const float*)d_in[5];
  const float* dwc_b   = (const float*)d_in[6];
  const float* pos_enc = (const float*)d_in[7];
  float* out = (float*)d_out;

  bf16* qs = (bf16*)d_ws;        // [12288][196][32] bf16
  bf16* ks = qs + QE;
  bf16* vs = ks + QE;
  bf16* os = qs;                 // attention+conv output aliases q (q is fully
                                 // consumed into LDS per-block before os writes)

  dim3 g1(1568, 9);
  qkv_gemm_kernel<<<g1, 256, 0, stream>>>(x, qkv_w, qkv_b, pos_enc, qs, ks, vs);

  attn_kernel<<<NBH, 128, 0, stream>>>(qs, ks, vs, dwc_w, dwc_b, os);

  dim3 g3(1568, 3);
  proj_gemm_kernel<<<g3, 256, 0, stream>>>(os, proj_w, proj_b, out);
}
